// SparseRNN_31860067401969
// MI455X (gfx1250) — matrix operand order repaired and checked by reading, not compiled, not run on hardware
//
#include <hip/hip_runtime.h>
#include <hip/hip_bf16.h>

// ---------------------------------------------------------------------------
// Sparse RNN on MI455X (gfx1250), densified to bf16 WMMA.
//
// - Densify COO weights to dense bf16 (atomic pk_add_bf16 coalesces dups).
//   Whh (128MiB) + Who (32MiB) stay L2-resident (192MB L2) across all 32
//   serial recurrence steps, so the recurrence is L2-BW bound, not HBM bound.
// - x is converted once to batch-major bf16 xb[t][n][k]; the time-parallel
//   input projection ihpre[t] = Wih @ x_t is one big batched WMMA GEMM.
// - Each wave computes a 16x32 output tile (two v_wmma_f32_16x16x32_bf16 per
//   32-wide K step), software-pipelined: next K-step fragments are loaded
//   before the current WMMAs so loads overlap matrix work.
// - Fragment layouts follow ISA 7.12.2 exactly:
//     A 16x32: lanes 0-15 K{0..7,16..23}, lanes 16-31 K{8..15,24..31}
//     B 32x16: lanes 0-15 K{0..7,16..23}, lanes 16-31 K{8..15,24..31}
//     C/D:     VGPR r -> row r (lanes 0-15) / row r+8 (lanes 16-31)
// ---------------------------------------------------------------------------

typedef __attribute__((ext_vector_type(16))) __bf16 v16bf;
typedef __attribute__((ext_vector_type(8)))  float  v8f;

constexpr int kB = 32, kT = 32, kI = 8192, kH = 8192, kO = 2048;

// workspace layout (bytes) — total ~337 MiB
constexpr size_t SZ_WIH = (size_t)kH * kI * 2;          // 128 MiB bf16
constexpr size_t SZ_WHH = (size_t)kH * kH * 2;          // 128 MiB bf16
constexpr size_t SZ_WHO = (size_t)kO * kH * 2;          //  32 MiB bf16
constexpr size_t SZ_HT  = (size_t)kB * kH * 2;          // 512 KiB bf16
constexpr size_t SZ_XB  = (size_t)kT * kB * kI * 2;     //  16 MiB bf16
constexpr size_t OFF_WIH   = 0;
constexpr size_t OFF_WHH   = OFF_WIH + SZ_WIH;
constexpr size_t OFF_WHO   = OFF_WHH + SZ_WHH;
constexpr size_t OFF_HTA   = OFF_WHO + SZ_WHO;
constexpr size_t OFF_HTB   = OFF_HTA + SZ_HT;
constexpr size_t OFF_XB    = OFF_HTB + SZ_HT;
constexpr size_t OFF_IHPRE = OFF_XB + SZ_XB;            // f32 [t][n][m], 32 MiB
constexpr size_t ZERO_BYTES = OFF_XB;                   // weights + both hT

// ---------------- helpers ----------------

__device__ __forceinline__ unsigned short f2bf_bits(float f) {
    union { float f; unsigned u; } v; v.f = f;
    unsigned r = v.u + 0x7FFFu + ((v.u >> 16) & 1u);    // round-to-nearest-even
    return (unsigned short)(r >> 16);
}

__device__ __forceinline__ v8f wmma_bf16(v16bf a, v16bf b, v8f c) {
    // (neg_a, A, neg_b, B, c_mod, C, reuse_a, reuse_b)
    return __builtin_amdgcn_wmma_f32_16x16x32_bf16(false, a, false, b,
                                                   (short)0, c, false, false);
}

// A fragment, 16x32 bf16 (MxK), row-major W with leading dim ld.
__device__ __forceinline__ v16bf load_a_bf16(const __bf16* W, int m_base,
                                             int k_base, int ld, int lane) {
    int m = m_base + (lane & 15);
    const __bf16* p = W + (size_t)m * ld + k_base + ((lane & 16) ? 8 : 0);
    union { uint4 q[2]; v16bf v; } u;
    u.q[0] = ((const uint4*)p)[0];   // 8 bf16 = 16B
    u.q[1] = ((const uint4*)p)[2];   // +16 elements
    return u.v;
}

// B fragment, 32x16 bf16 (KxN), from batch-major X[n][k] with leading dim ld.
__device__ __forceinline__ v16bf load_b_bf16(const __bf16* X, int n_base,
                                             int k_base, int ld, int lane) {
    int n = n_base + (lane & 15);
    const __bf16* p = X + (size_t)n * ld + k_base + ((lane & 16) ? 8 : 0);
    union { uint4 q[2]; v16bf v; } u;
    u.q[0] = ((const uint4*)p)[0];
    u.q[1] = ((const uint4*)p)[2];
    return u.v;
}

// ---------------- kernels ----------------

__global__ void zero_kernel(uint4* p, long n16) {
    long i = (long)blockIdx.x * blockDim.x + threadIdx.x;
    uint4 z = {0u, 0u, 0u, 0u};
    long stride = (long)gridDim.x * blockDim.x;
    for (; i < n16; i += stride) p[i] = z;
}

// One-time layout change + precision cvt: x (B,T,I) f32 -> xb[t][n][k] bf16.
// Each thread converts 8 contiguous k (two float4 loads -> one uint4 store).
__global__ void cvt_x(const float* __restrict__ x, __bf16* __restrict__ xb) {
    int idx = blockIdx.x * blockDim.x + threadIdx.x;   // 1,048,576 threads
    int idx8 = idx << 3;
    int k   = idx8 & (kI - 1);
    int row = idx8 >> 13;          // dest row = t*kB + n
    int t = row >> 5;
    int n = row & 31;
    const float* s = x + ((size_t)n * kT + t) * kI + k;
    union { unsigned short h[8]; uint4 q; } o;
#pragma unroll
    for (int i = 0; i < 8; ++i) o.h[i] = f2bf_bits(s[i]);
    *(uint4*)(xb + (size_t)row * kI + k) = o.q;
}

// COO scatter with duplicate coalescing: packed-bf16 atomic add into dense W.
__global__ void scatter_coo(const int* __restrict__ rows,
                            const int* __restrict__ cols,
                            const float* __restrict__ vals,
                            int nnz, __bf16* W, int ld) {
    int i = blockIdx.x * blockDim.x + threadIdx.x;
    if (i >= nnz) return;
    size_t idx = (size_t)rows[i] * ld + (size_t)cols[i];
    unsigned short hv = f2bf_bits(vals[i]);
    unsigned packed = (idx & 1) ? ((unsigned)hv << 16) : (unsigned)hv;  // other half +0.0
    unsigned long long addr = (unsigned long long)W + (idx >> 1) * 4ull;
    asm volatile("global_atomic_pk_add_bf16 %0, %1, off"
                 :: "v"(addr), "v"(packed) : "memory");
}

// Software-pipelined 16x32-tile GEMM body: C += W[m_base:,:K] @ X^T, X batch-major.
__device__ __forceinline__ void gemm_tile(const __bf16* __restrict__ W,
                                          const __bf16* __restrict__ X,
                                          int m_base, int K, int lane,
                                          v8f& c0, v8f& c1) {
    v16bf a  = load_a_bf16(W, m_base, 0, K, lane);
    v16bf b0 = load_b_bf16(X, 0, 0, K, lane);
    v16bf b1 = load_b_bf16(X, 16, 0, K, lane);
    for (int k = 32; k < K; k += 32) {
        v16bf an  = load_a_bf16(W, m_base, k, K, lane);   // next-step loads issue
        v16bf b0n = load_b_bf16(X, 0, k, K, lane);        // before current WMMAs
        v16bf b1n = load_b_bf16(X, 16, k, K, lane);
        c0 = wmma_bf16(a, b0, c0);
        c1 = wmma_bf16(a, b1, c1);
        a = an; b0 = b0n; b1 = b1n;
    }
    c0 = wmma_bf16(a, b0, c0);
    c1 = wmma_bf16(a, b1, c1);
}

// ihpre[t][n][m] = (Wih @ x_t)[m][n] for all t — time-parallel, 16384 waves.
__global__ void ihpre_gemm(const __bf16* __restrict__ Wih,
                           const __bf16* __restrict__ xb,
                           float* __restrict__ ihpre) {
    int lane = threadIdx.x & 31;
    int wave = (int)((blockIdx.x * blockDim.x + threadIdx.x) >> 5);
    int t = wave >> 9;                 // /512 m-tiles
    int m_base = (wave & 511) * 16;
    v8f c0 = {}, c1 = {};
    gemm_tile(Wih, xb + (size_t)t * kB * kI, m_base, kI, lane, c0, c1);
    int col = lane & 15;
    int m0 = m_base + ((lane & 16) ? 8 : 0);
    float* p0 = ihpre + ((size_t)(t * kB + col) * kH) + m0;
    float* p1 = ihpre + ((size_t)(t * kB + 16 + col) * kH) + m0;
#pragma unroll
    for (int r = 0; r < 8; ++r) { p0[r] = c0[r]; p1[r] = c1[r]; }
}

// h_new = sigmoid(Whh @ h + ihpre[t] + bias); 512 waves, Whh L2-resident.
__global__ void step_hh(const __bf16* __restrict__ Whh,
                        const __bf16* __restrict__ hin,
                        const float* __restrict__ ihpre,
                        const float* __restrict__ bias,
                        __bf16* __restrict__ hout, int t) {
    int lane = threadIdx.x & 31;
    int wave = (int)((blockIdx.x * blockDim.x + threadIdx.x) >> 5);
    int m_base = wave * 16;
    int col = lane & 15;
    int m0 = m_base + ((lane & 16) ? 8 : 0);
    v8f c0, c1;
    const float* p0 = ihpre + ((size_t)(t * kB + col) * kH) + m0;
    const float* p1 = ihpre + ((size_t)(t * kB + 16 + col) * kH) + m0;
#pragma unroll
    for (int r = 0; r < 8; ++r) {
        float bb = bias[m0 + r];
        c0[r] = p0[r] + bb;
        c1[r] = p1[r] + bb;
    }
    gemm_tile(Whh, hin, m_base, kH, lane, c0, c1);
    // sigmoid + store batch-major bf16 (8 contiguous elems = one 16B store)
    union { unsigned short s[8]; uint4 q; } o0, o1;
#pragma unroll
    for (int r = 0; r < 8; ++r) {
        o0.s[r] = f2bf_bits(1.0f / (1.0f + __expf(-c0[r])));
        o1.s[r] = f2bf_bits(1.0f / (1.0f + __expf(-c1[r])));
    }
    *(uint4*)(hout + (size_t)col * kH + m0)        = o0.q;
    *(uint4*)(hout + (size_t)(16 + col) * kH + m0) = o1.q;
}

// out[b][t][o] = (Who @ h_new)[o][b] + bias[o]; 128 waves, Who L2-resident.
__global__ void step_ho(const __bf16* __restrict__ Who,
                        const __bf16* __restrict__ hin,
                        const float* __restrict__ bias,
                        float* __restrict__ out, int t) {
    int lane = threadIdx.x & 31;
    int wave = (int)((blockIdx.x * blockDim.x + threadIdx.x) >> 5);
    int m_base = wave * 16;
    int col = lane & 15;
    int m0 = m_base + ((lane & 16) ? 8 : 0);
    v8f c0, c1;
#pragma unroll
    for (int r = 0; r < 8; ++r) { float bb = bias[m0 + r]; c0[r] = bb; c1[r] = bb; }
    gemm_tile(Who, hin, m_base, kH, lane, c0, c1);
    float* q0 = out + ((size_t)col * kT + t) * kO + m0;
    float* q1 = out + ((size_t)(16 + col) * kT + t) * kO + m0;
#pragma unroll
    for (int r = 0; r < 8; ++r) { q0[r] = c0[r]; q1[r] = c1[r]; }
}

// ---------------- launch ----------------

extern "C" void kernel_launch(void* const* d_in, const int* in_sizes, int n_in,
                              void* d_out, int out_size, void* d_ws, size_t ws_size,
                              hipStream_t stream) {
    const float* x       = (const float*)d_in[0];
    const int*   hh_rows = (const int*)d_in[1];
    const int*   hh_cols = (const int*)d_in[2];
    const float* hh_vals = (const float*)d_in[3];
    const float* hh_bias = (const float*)d_in[4];
    const int*   ih_rows = (const int*)d_in[5];
    const int*   ih_cols = (const int*)d_in[6];
    const float* ih_vals = (const float*)d_in[7];
    const int*   ho_rows = (const int*)d_in[8];
    const int*   ho_cols = (const int*)d_in[9];
    const float* ho_vals = (const float*)d_in[10];
    const float* ho_bias = (const float*)d_in[11];
    float* out = (float*)d_out;

    char* ws = (char*)d_ws;   // requires ~337 MiB (see layout constants)
    __bf16* Wih = (__bf16*)(ws + OFF_WIH);
    __bf16* Whh = (__bf16*)(ws + OFF_WHH);
    __bf16* Who = (__bf16*)(ws + OFF_WHO);
    __bf16* hTa = (__bf16*)(ws + OFF_HTA);
    __bf16* hTb = (__bf16*)(ws + OFF_HTB);
    __bf16* xb  = (__bf16*)(ws + OFF_XB);
    float*  ihp = (float*) (ws + OFF_IHPRE);

    int nnz_hh = in_sizes[1], nnz_ih = in_sizes[5], nnz_ho = in_sizes[8];

    // 1) zero dense weights + both hidden-state buffers (h0 = 0) every call
    zero_kernel<<<4096, 256, 0, stream>>>((uint4*)ws, (long)(ZERO_BYTES / 16));

    // 2) one-time x layout/precision conversion (f32 -> batch-major bf16)
    cvt_x<<<4096, 256, 0, stream>>>(x, xb);

    // 3) densify COO -> bf16 (duplicates coalesce via packed-bf16 atomic add)
    scatter_coo<<<(nnz_ih + 255) / 256, 256, 0, stream>>>(ih_rows, ih_cols, ih_vals, nnz_ih, Wih, kI);
    scatter_coo<<<(nnz_hh + 255) / 256, 256, 0, stream>>>(hh_rows, hh_cols, hh_vals, nnz_hh, Whh, kH);
    scatter_coo<<<(nnz_ho + 255) / 256, 256, 0, stream>>>(ho_rows, ho_cols, ho_vals, nnz_ho, Who, kH);

    // 4) time-parallel input projection: 512 m-tiles x 32 t = 16384 waves
    ihpre_gemm<<<2048, 256, 0, stream>>>(Wih, xb, ihp);

    // 5) serial recurrence: Whh (128MiB) + Who (32MiB) stay in 192MB L2
    for (int t = 0; t < kT; ++t) {
        const __bf16* hin = (t & 1) ? hTb : hTa;
        __bf16*      hout = (t & 1) ? hTa : hTb;
        step_hh<<<64, 256, 0, stream>>>(Whh, hin, ihp, hh_bias, hout, t);
        step_ho<<<16, 256, 0, stream>>>(Who, hout, ho_bias, out, t);
    }
    (void)n_in; (void)out_size; (void)ws_size;
}
